// BiLSTM_12154757447891
// MI455X (gfx1250) — matrix operand order repaired and checked by reading, not compiled, not run on hardware
//
#include <hip/hip_runtime.h>
#include <hip/hip_bf16.h>
#include <stdint.h>

// ---------------------------------------------------------------------------
// BiLSTM (B=32, L=1024, D=H=512) persistent-kernel implementation for gfx1250.
//   - bf16 WMMA 16x16x32, f32 accumulate; c-state lives in VGPRs.
//   - x_t / h_{t-1} tiles staged to LDS via TDM tensor_load_to_lds (guarded).
//   - per-direction device-scope generation barrier between time steps.
//   - step loop templated on direction: fwd path has no reset-mask VALU.
// ---------------------------------------------------------------------------

typedef __bf16 bf16_t;
typedef __bf16        v16bf __attribute__((ext_vector_type(16)));
typedef float         v8f   __attribute__((ext_vector_type(8)));
typedef unsigned int  u32x4 __attribute__((ext_vector_type(4)));
typedef int           i32x4 __attribute__((ext_vector_type(4)));
typedef int           i32x8 __attribute__((ext_vector_type(8)));

#define B_   32
#define L_   1024
#define D_   512
#define H_   512
#define N4H  2048       // 4*H
#define KT   32         // k-tiles of 32 over K = D+H = 1024
#define NT   128        // n-tiles of 16 over 4H = 2048

// workspace layout (bytes)
#define SYNC_OFF   0                                  // 2 dirs x {count,gen}
#define HBUF_OFF   256                                // [2 dir][2 buf][32][512] bf16
#define HBUF_BYTES (2*2*32*512*2)
#define WFRAG_OFF  (1u<<19)                           // 512 KB
#define WFRAG_ELEMS_PER_DIR ((size_t)KT*NT*32*16)     // 2M bf16 = 4 MB
#define XBF_OFF    (16u<<20)                          // 16 MB
// total ws needed: 16MB + 32MB = 48MB

#if defined(__AMDGCN__) && __has_builtin(__builtin_amdgcn_tensor_load_to_lds)
#define USE_TDM 1
#else
#define USE_TDM 0
#endif

// ---------------------------------------------------------------------------
__global__ void bilstm_x_to_bf16(const float* __restrict__ x,
                                 bf16_t* __restrict__ xb, int n) {
  int i = blockIdx.x * blockDim.x + threadIdx.x;
  int stride = gridDim.x * blockDim.x;
  for (; i < n; i += stride) xb[i] = (bf16_t)x[i];
}

// Swizzle W[dir] (fp32 [1024,2048] row-major) into per-tile lane-major bf16
// B-fragments: frag[dir][kt][nt][lane][16].
// B-matrix (32x16) k order used here (A uses the same): lanes 0-15 hold col=lane, K=kt*32+0..15;
// lanes 16-31 hold col=lane-16, K=kt*32+16..31; element e of the 16-vector is K-offset e.
__global__ void bilstm_w_swizzle(const float* __restrict__ Wf,
                                 const float* __restrict__ Wb,
                                 bf16_t* __restrict__ frag) {
  int tid = blockIdx.x * blockDim.x + threadIdx.x;
  int total = 2 * KT * NT * 32;
  if (tid >= total) return;
  int lane = tid & 31;
  int nt   = (tid >> 5) % NT;
  int kt   = ((tid >> 5) / NT) % KT;
  int dir  = tid / (32 * NT * KT);
  const float* W = dir ? Wb : Wf;
  int col   = nt * 16 + (lane & 15);
  int kbase = kt * 32 + (lane >> 4) * 16;
  union { bf16_t h[16]; v16bf v; } o;
#pragma unroll
  for (int v = 0; v < 16; ++v)
    o.h[v] = (bf16_t)W[(size_t)(kbase + v) * N4H + col];
  bf16_t* dst = frag + (size_t)dir * WFRAG_ELEMS_PER_DIR
                     + (((size_t)kt * NT + nt) * 32 + lane) * 16;
  *(v16bf*)dst = o.v;
}

// ---------------------------------------------------------------------------
__device__ __forceinline__ float sigf(float x) {
  return 1.0f / (1.0f + __expf(-x));
}
__device__ __forceinline__ float tanh_fast(float x) {
  return 2.0f * sigf(2.0f * x) - 1.0f;   // NaN-free saturating tanh
}

#if USE_TDM
// 2D tile load (rows x width elems of 2-byte data) Global -> LDS via TDM.
__device__ __forceinline__ void tdm_load_2d_b16(unsigned lds_off, const void* gptr,
                                                unsigned width, unsigned rows,
                                                unsigned long long row_stride_elems) {
  unsigned long long ga = (unsigned long long)(uintptr_t)gptr;
  u32x4 g0;
  g0[0] = 1u;                                            // count=1, user D#
  g0[1] = lds_off;                                       // lds_addr (bytes)
  g0[2] = (unsigned)(ga & 0xffffffffu);                  // global_addr lo
  g0[3] = (unsigned)((ga >> 32) & 0x1ffffffu) | (2u << 30); // addr hi | type=2
  i32x8 g1;
  g1[0] = (int)(1u << 16);                               // data_size=1 (2B)
  g1[1] = (int)((width & 0xffffu) << 16);                // tensor_dim0 lo16
  g1[2] = (int)(((width >> 16) & 0xffffu) | ((rows & 0xffffu) << 16)); // dim0 hi | dim1 lo
  g1[3] = (int)(((rows >> 16) & 0xffffu) | ((width & 0xffffu) << 16)); // dim1 hi | tile_dim0
  g1[4] = (int)(rows & 0xffffu);                         // tile_dim1 (tile_dim2=0)
  g1[5] = (int)(row_stride_elems & 0xffffffffu);         // tensor_dim0_stride lo
  g1[6] = (int)((row_stride_elems >> 32) & 0xffffu);     // stride hi (dim1_stride unused)
  g1[7] = 0;
  i32x4 g2 = {0, 0, 0, 0}, g3 = {0, 0, 0, 0};
#if defined(__clang_major__) && (__clang_major__ >= 23)
  i32x8 g4 = {0, 0, 0, 0, 0, 0, 0, 0};
  __builtin_amdgcn_tensor_load_to_lds(g0, g1, g2, g3, g4, 0);
#else
  __builtin_amdgcn_tensor_load_to_lds(g0, g1, g2, g3, 0);
#endif
}
#endif

__device__ __forceinline__ void grid_sync(unsigned int* cnt, unsigned int* gen,
                                          unsigned int n) {
  __syncthreads();
  if (threadIdx.x == 0) {
    unsigned int g = __hip_atomic_load(gen, __ATOMIC_ACQUIRE, __HIP_MEMORY_SCOPE_AGENT);
    unsigned int old = __hip_atomic_fetch_add(cnt, 1u, __ATOMIC_ACQ_REL, __HIP_MEMORY_SCOPE_AGENT);
    if (old == n - 1u) {
      __hip_atomic_store(cnt, 0u, __ATOMIC_RELAXED, __HIP_MEMORY_SCOPE_AGENT);
      __hip_atomic_fetch_add(gen, 1u, __ATOMIC_RELEASE, __HIP_MEMORY_SCOPE_AGENT);
    } else {
      while (__hip_atomic_load(gen, __ATOMIC_ACQUIRE, __HIP_MEMORY_SCOPE_AGENT) == g)
        __builtin_amdgcn_s_sleep(2);
    }
  }
  __syncthreads();
}

// ---------------------------------------------------------------------------
// Time-step loop, specialized per direction (DIR=0 fwd, DIR=1 bwd+ResetCore).
template <int DIR>
__device__ __forceinline__ void lstm_steps(
    unsigned char* sm, const int* lsh,
    const bf16_t* __restrict__ xbf, const bf16_t* __restrict__ wdir,
    float* __restrict__ out, bf16_t* __restrict__ hbuf,
    unsigned int* cnt, unsigned int* gen,
    int b0, int htile, int lane, const float bv[4]) {
  const int row  = lane & 15;
  const int half = lane >> 4;
  const unsigned rowoff = (unsigned)row * 1024u;        // 512 bf16 per row

  // hoist lengths into registers (constants for the whole scan)
  int len_frag = 0x7fffffff;
  int lenc[8];
#pragma unroll
  for (int r = 0; r < 8; ++r) lenc[r] = 0x7fffffff;
  if (DIR) {
    len_frag = lsh[b0 + row];
#pragma unroll
    for (int r = 0; r < 8; ++r) lenc[r] = lsh[b0 + half * 8 + r];
  }

  v8f cst = {0.f, 0.f, 0.f, 0.f, 0.f, 0.f, 0.f, 0.f};  // cell state, resident

  for (int s = 0; s < L_; ++s) {
    const int t = DIR ? (L_ - 1 - s) : s;

    // ---- stage x_t rows [b0..b0+15] and h_{s-1} rows into LDS -------------
#if USE_TDM
    if (threadIdx.x < 32) {
      unsigned smb = (unsigned)(uintptr_t)(void*)sm;    // low 32 bits = LDS offset
      tdm_load_2d_b16(smb,
                      xbf + ((size_t)b0 * L_ + t) * D_,
                      512u, 16u, (unsigned long long)L_ * D_);
      tdm_load_2d_b16(smb + 16384u,
                      hbuf + (((size_t)(DIR * 2 + (s & 1)) * 32 + b0) * 512),
                      512u, 16u, 512ull);
      __builtin_amdgcn_s_wait_tensorcnt(0);
    }
    __syncthreads();
#else
    for (int c = threadIdx.x; c < 1024; c += 256) {
      int r = c >> 6, e = (c & 63) * 8;
      *(uint4*)(sm + r * 1024 + (c & 63) * 16) =
          *(const uint4*)(xbf + (((size_t)(b0 + r) * L_ + t) * D_) + e);
    }
    for (int c = threadIdx.x; c < 1024; c += 256) {
      int r = c >> 6, e = (c & 63) * 8;
      *(uint4*)(sm + 16384 + r * 1024 + (c & 63) * 16) =
          *(const uint4*)(hbuf + (((size_t)(DIR * 2 + (s & 1)) * 32 + b0 + r) * 512) + e);
    }
    __syncthreads();
#endif

    // ResetCore (bwd only): zero h,c BEFORE the step where t >= len-1
    const bool keep_lane = DIR ? (t < len_frag - 1) : true;

    // ---- 4 gate-tile GEMMs over K = 1024 ----------------------------------
    v8f acc[4];
#pragma unroll
    for (int g = 0; g < 4; ++g) {
#pragma unroll
      for (int r = 0; r < 8; ++r) acc[g][r] = bv[g];
    }

#pragma unroll 4
    for (int kt = 0; kt < KT; ++kt) {
      const unsigned blk = (kt < 16) ? 0u : 16384u;
      const unsigned ktl = (kt < 16) ? (unsigned)kt : (unsigned)(kt - 16);
      const unsigned boff = blk + rowoff + ktl * 64u + (unsigned)half * 32u;
      union { uint4 u[2]; v16bf v; } A;
      A.u[0] = *(const uint4*)(sm + boff);
      A.u[1] = *(const uint4*)(sm + boff + 16);
      if (DIR && kt >= 16 && !keep_lane) {     // mask h rows for reset batches
        uint4 z = {0u, 0u, 0u, 0u};
        A.u[0] = z; A.u[1] = z;
      }
      const bf16_t* wbase = wdir + (((size_t)kt * NT) * 32 + lane) * 16;
#pragma unroll
      for (int g = 0; g < 4; ++g) {
        const v16bf bfr = *(const v16bf*)(wbase + (size_t)(g * 32 + htile) * 512);
        acc[g] = __builtin_amdgcn_wmma_f32_16x16x32_bf16(
            false, A.v, false, bfr, (short)0, acc[g], false, false);
      }
    }

    // ---- cell update + stores --------------------------------------------
#pragma unroll
    for (int r = 0; r < 8; ++r) {
      const int m = half * 8 + r;              // M row of this component
      const int brow = b0 + m;
      float cprev = cst[r];
      if (DIR && (t >= lenc[r] - 1)) cprev = 0.0f;
      const float iv = acc[0][r], gv = acc[1][r], fv = acc[2][r], ov = acc[3][r];
      const float cnew = sigf(fv + 1.0f) * cprev + sigf(iv) * tanh_fast(gv);
      const float hnew = sigf(ov) * tanh_fast(cnew);
      cst[r] = cnew;
      const int j = htile * 16 + (lane & 15);
      out[((size_t)brow * L_ + t) * (2 * H_) + (size_t)DIR * H_ + j] = hnew;
      hbuf[(((size_t)(DIR * 2 + ((s + 1) & 1)) * 32 + brow) * 512) + j] = (bf16_t)hnew;
    }

    __threadfence();
    grid_sync(cnt, gen, 8u);                   // 8 WGs per direction
  }
}

// ---------------------------------------------------------------------------
// Persistent recurrent kernel: 16 WGs x 256 threads (8 waves).
//   wg: dir = wg>>3; sub = wg&7; mtile = sub>>2 (16 batch rows); htile = (sub&3)*8 + wave.
// Each wave computes 4 gate tiles (i,g,f,o) of 16x16 and does the cell update.
__global__ void __launch_bounds__(256, 1)
bilstm_recurrent(const bf16_t* __restrict__ xbf,
                 const bf16_t* __restrict__ wfrag,
                 const float* __restrict__ b_fwd,
                 const float* __restrict__ b_bwd,
                 const int* __restrict__ lengths,
                 float* __restrict__ out,
                 bf16_t* __restrict__ hbuf,
                 unsigned int* __restrict__ syncv) {
  __shared__ __align__(128) unsigned char sm[33024];   // 16KB x | 16KB h | 32 ints len
  int* lsh = (int*)(sm + 32768);

  const int wg     = blockIdx.x;
  const int dir    = wg >> 3;
  const int sub    = wg & 7;
  const int mtile  = sub >> 2;
  const int waveId = threadIdx.x >> 5;
  const int lane   = threadIdx.x & 31;
  const int htile  = (sub & 3) * 8 + waveId;
  const int b0     = mtile * 16;

  if (threadIdx.x < 32) lsh[threadIdx.x] = lengths[threadIdx.x];
  __syncthreads();

  const float* bias = dir ? b_bwd : b_fwd;
  float bv[4];
#pragma unroll
  for (int g = 0; g < 4; ++g)
    bv[g] = bias[g * H_ + htile * 16 + (lane & 15)];

  const bf16_t* wdir = wfrag + (size_t)dir * WFRAG_ELEMS_PER_DIR;
  unsigned int* cnt = syncv + dir * 2;
  unsigned int* gen = syncv + dir * 2 + 1;

  if (dir == 0)
    lstm_steps<0>(sm, lsh, xbf, wdir, out, hbuf, cnt, gen, b0, htile, lane, bv);
  else
    lstm_steps<1>(sm, lsh, xbf, wdir, out, hbuf, cnt, gen, b0, htile, lane, bv);
}

// ---------------------------------------------------------------------------
extern "C" void kernel_launch(void* const* d_in, const int* in_sizes, int n_in,
                              void* d_out, int out_size, void* d_ws, size_t ws_size,
                              hipStream_t stream) {
  const float* x     = (const float*)d_in[0];
  const float* W_fwd = (const float*)d_in[1];
  const float* b_fwd = (const float*)d_in[2];
  const float* W_bwd = (const float*)d_in[3];
  const float* b_bwd = (const float*)d_in[4];
  const int*   lens  = (const int*)d_in[5];

  char* ws = (char*)d_ws;
  unsigned int* syncv = (unsigned int*)(ws + SYNC_OFF);
  bf16_t* hbuf  = (bf16_t*)(ws + HBUF_OFF);
  bf16_t* wfrag = (bf16_t*)(ws + WFRAG_OFF);
  bf16_t* xbf   = (bf16_t*)(ws + XBF_OFF);

  // zero barrier counters + h double-buffers (deterministic per launch)
  hipMemsetAsync(d_ws, 0, HBUF_OFF + HBUF_BYTES, stream);

  bilstm_x_to_bf16<<<2048, 256, 0, stream>>>(x, xbf, B_ * L_ * D_);
  bilstm_w_swizzle<<<(2 * KT * NT * 32 + 255) / 256, 256, 0, stream>>>(W_fwd, W_bwd, wfrag);
  bilstm_recurrent<<<16, 256, 0, stream>>>(xbf, wfrag, b_fwd, b_bwd, lens,
                                           (float*)d_out, hbuf, syncv);
}
